// MultiHeadSelfAttention_33732673143663
// MI455X (gfx1250) — matrix operand order repaired and checked by reading, not compiled, not run on hardware
//
#include <hip/hip_runtime.h>
#include <hip/hip_bf16.h>

typedef __attribute__((ext_vector_type(16))) __bf16 v16bf;
typedef __attribute__((ext_vector_type(8)))  __bf16 v8bf;
typedef __attribute__((ext_vector_type(4)))  __bf16 v4bf;
typedef __attribute__((ext_vector_type(8)))  float  v8f;
typedef __attribute__((ext_vector_type(4)))  int    v4i;
typedef __attribute__((ext_vector_type(8)))  int    v8i;
typedef __attribute__((ext_vector_type(4)))  unsigned int v4u;

// ---------------- CDNA5 helpers ----------------

#if __has_builtin(__builtin_amdgcn_global_load_async_to_lds_b128)
#define HAVE_ASYNC_LDS 1
#endif

#if __has_builtin(__builtin_amdgcn_tensor_load_to_lds)
#define HAVE_TDM 1
#endif

__device__ __forceinline__ void cp16_g2l(__bf16* lds, const __bf16* g) {
#ifdef HAVE_ASYNC_LDS
  __builtin_amdgcn_global_load_async_to_lds_b128(
      (__attribute__((address_space(1))) v4i*)g,
      (__attribute__((address_space(3))) v4i*)lds, 0, 0);
#else
  *(v8bf*)lds = *(const v8bf*)g;   // fallback: global_load_b128 + ds_store_b128
#endif
}

__device__ __forceinline__ void cp_wait() {
#ifdef HAVE_ASYNC_LDS
#if __has_builtin(__builtin_amdgcn_s_wait_asynccnt)
  __builtin_amdgcn_s_wait_asynccnt(0);
#else
  asm volatile("s_wait_asynccnt 0" ::: "memory");
#endif
#endif
}

__device__ __forceinline__ unsigned lds_off(const void* p) {
  return (unsigned)(unsigned long long)
      (__attribute__((address_space(3))) const void*)p;
}

#ifdef HAVE_TDM
// Tensor DMA: load a (rows=128) x (32 bf16) tile, row stride = ld elems, into
// LDS with 16B pad after each 64B row chunk -> padded row stride 80B (LDSP=40).
__device__ __forceinline__ void tdm_load_tile(unsigned ldsOff, const __bf16* g,
                                              int ldElems) {
  unsigned long long ga = (unsigned long long)(size_t)g;
  v4u g0;
  g0[0] = 1u;                                       // count=1 (valid descriptor)
  g0[1] = ldsOff;                                   // lds_addr
  g0[2] = (unsigned)(ga & 0xFFFFFFFFull);           // global_addr[31:0]
  g0[3] = (unsigned)((ga >> 32) & 0x01FFFFFFull)    // global_addr[56:32]
        | (2u << 30);                               // type = 2 ("image")
  v8i g1;
  g1[0] = (1 << 16)     // data_size = 1 -> 2 bytes
        | (1 << 20)     // pad_enable
        | (3 << 22)     // pad_interval: 16 DWORDs (64B)
        | (3 << 25);    // pad_amount:   4 DWORDs (16B)
  g1[1] = (int)(0u << 16);                 // barrier_addr=0 | tensor_dim0 lo16
  g1[2] = (int)(0x4000u | (0u << 16));     // tensor_dim0 hi16 (2^30) | dim1 lo16
  g1[3] = (int)(0x4000u | (32u << 16));    // tensor_dim1 hi16 (2^30) | tile_dim0=32
  g1[4] = 128;                             // tile_dim1 = 128 rows, tile_dim2 = 0
  g1[5] = ldElems;                         // tensor_dim0_stride lo32 (elems)
  g1[6] = 0;                               // stride hi16 | dim1_stride lo16
  g1[7] = 0;
  __builtin_amdgcn_tensor_load_to_lds(g0, g1, (v4i){0,0,0,0}, (v4i){0,0,0,0},
                                      (v8i){0,0,0,0,0,0,0,0}, 0);
}
#endif

__device__ __forceinline__ void tile_wait() {
#ifdef HAVE_TDM
#if __has_builtin(__builtin_amdgcn_s_wait_tensorcnt)
  __builtin_amdgcn_s_wait_tensorcnt(0);
#else
  asm volatile("s_wait_tensorcnt 0" ::: "memory");
#endif
#else
  cp_wait();
#endif
}

__device__ __forceinline__ v16bf cat88(v8bf lo, v8bf hi) {
  return __builtin_shufflevector(lo, hi, 0,1,2,3,4,5,6,7,8,9,10,11,12,13,14,15);
}

__device__ __forceinline__ v8f wmma_bf16(v16bf a, v16bf b, v8f c) {
  return __builtin_amdgcn_wmma_f32_16x16x32_bf16(false, a, false, b, (short)0, c,
                                                 false, false);
}

// ---------------- fp32 -> bf16 convert ----------------

__global__ __launch_bounds__(256) void cvt_f32_bf16(
    const float* __restrict__ in, __bf16* __restrict__ out, int n4) {
  int i = blockIdx.x * blockDim.x + threadIdx.x;
  int st = gridDim.x * blockDim.x;
  for (; i < n4; i += st) {
    float4 v = ((const float4*)in)[i];
    v4bf o;
    o[0] = (__bf16)v.x; o[1] = (__bf16)v.y; o[2] = (__bf16)v.z; o[3] = (__bf16)v.w;
    ((v4bf*)out)[i] = o;
  }
}

// ---------------- generic NT GEMM: C[m,n] = sum_k A[m,k]*Bm[n,k] (+bias[n]) ----
// MODE 0: bf16 row-major out, 1: bf16 transposed out, 2: f32 row-major out.
// Block tile 128x128, KSTEP 32, 8 waves each computing a 32x64 macro-tile
// (2x4 grid of 16x16 WMMA tiles). Double-buffered LDS; tiles staged by the
// Tensor Data Mover (wave 0 issues one tensor_load_to_lds per matrix per tile).

#define BT 128
#define KST 32
#define LDSP 40   // 32 + 8 pad elems -> conflict-free 16B fragment loads

template<int MODE>
__global__ __launch_bounds__(256) void gemm_nt_bf16(
    const __bf16* __restrict__ A, const __bf16* __restrict__ Bm,
    const float* __restrict__ bias, void* __restrict__ Cout,
    int Kdim, int lda, int ldb, int ldc,
    long long sA, long long sB, long long sC) {
  __shared__ __attribute__((aligned(16))) __bf16 shA[2][BT * LDSP];
  __shared__ __attribute__((aligned(16))) __bf16 shB[2][BT * LDSP];

  const int tid  = threadIdx.x;
  const int lane = tid & 31;
  const int w    = tid >> 5;
  const int wm   = w >> 1;                 // 0..3
  const int wn   = w & 1;                  // 0..1
  const int z    = blockIdx.z;
  A  += (long long)z * sA;
  Bm += (long long)z * sB;
  const int mBase = blockIdx.y * BT;
  const int nBase = blockIdx.x * BT;

  const int l15    = lane & 15;
  const int hi     = lane >> 4;
  const int klo8   = hi << 3;              // A frag k offset (0 / 8)
  const int koff16 = hi << 4;              // unused; B frags use klo8 like A

  auto load_tile = [&](int buf, int k0) {
#ifdef HAVE_TDM
    if (w == 0) {                          // one wave drives the TDM
      tdm_load_tile(lds_off(&shA[buf][0]), A  + (size_t)mBase * lda + k0, lda);
      tdm_load_tile(lds_off(&shB[buf][0]), Bm + (size_t)nBase * ldb + k0, ldb);
    }
#else
#pragma unroll
    for (int i = 0; i < 2; ++i) {          // 512 16B chunks per matrix / 256 thr
      int c   = tid + (i << 8);
      int r   = c >> 2;                    // 4 chunks per 32-elem row
      int off = (c & 3) << 3;
      cp16_g2l(&shA[buf][r * LDSP + off], A  + (size_t)(mBase + r) * lda + k0 + off);
      cp16_g2l(&shB[buf][r * LDSP + off], Bm + (size_t)(nBase + r) * ldb + k0 + off);
    }
#endif
  };

  v8f acc[2][4];
#pragma unroll
  for (int mi = 0; mi < 2; ++mi)
#pragma unroll
    for (int ni = 0; ni < 4; ++ni) acc[mi][ni] = (v8f){0,0,0,0,0,0,0,0};

  const int nk = Kdim / KST;
  load_tile(0, 0);
  tile_wait();
  __syncthreads();

  for (int kt = 0; kt < nk; ++kt) {
    int cur = kt & 1;
    if (kt + 1 < nk) load_tile(cur ^ 1, (kt + 1) * KST);

    v16bf af[2], bfrag[4];
#pragma unroll
    for (int mi = 0; mi < 2; ++mi) {
      const __bf16* p = &shA[cur][(wm * 32 + mi * 16 + l15) * LDSP + klo8];
      af[mi] = cat88(*(const v8bf*)p, *(const v8bf*)(p + 16));
    }
#pragma unroll
    for (int ni = 0; ni < 4; ++ni) {
      const __bf16* p = &shB[cur][(wn * 64 + ni * 16 + l15) * LDSP + klo8];
      bfrag[ni] = cat88(*(const v8bf*)p, *(const v8bf*)(p + 16));
    }
#pragma unroll
    for (int mi = 0; mi < 2; ++mi)
#pragma unroll
      for (int ni = 0; ni < 4; ++ni)
        acc[mi][ni] = wmma_bf16(af[mi], bfrag[ni], acc[mi][ni]);

    tile_wait();
    __syncthreads();
  }

  // epilogue: C layout = VGPR r holds row (r + 8*hi), col = l15
#pragma unroll
  for (int mi = 0; mi < 2; ++mi) {
#pragma unroll
    for (int ni = 0; ni < 4; ++ni) {
      int n  = nBase + wn * 64 + ni * 16 + l15;
      int m0 = mBase + wm * 32 + mi * 16 + (hi << 3);
      float bi = bias ? bias[n] : 0.0f;
#pragma unroll
      for (int r = 0; r < 8; ++r) {
        float v = acc[mi][ni][r] + bi;
        if (MODE == 0)
          ((__bf16*)Cout)[(size_t)z * sC + (size_t)(m0 + r) * ldc + n] = (__bf16)v;
        else if (MODE == 1)
          ((__bf16*)Cout)[(size_t)z * sC + (size_t)n * ldc + (m0 + r)] = (__bf16)v;
        else
          ((float*)Cout)[(size_t)z * sC + (size_t)(m0 + r) * ldc + n] = v;
      }
    }
  }
}

// ---------------- fused scores + softmax ----------------
// One workgroup per (batch, 16 query rows). Q tile in LDS (async-LDS staged);
// each of 8 waves computes 16 s-tiles of Q·K^T (K frags read directly,
// L2-resident), keeps the full 16x2048 score block in registers, softmaxes,
// writes bf16 probs.

#define QP 1032   // 1024 + 8 pad

__global__ __launch_bounds__(256) void attn_scores_softmax(
    const __bf16* __restrict__ Q, const __bf16* __restrict__ Km,
    __bf16* __restrict__ P, float scale) {
  __shared__ __attribute__((aligned(16))) __bf16 sQ[16 * QP];
  __shared__ float sredM[8][16];
  __shared__ float sredS[8][16];

  const int tid  = threadIdx.x;
  const int lane = tid & 31;
  const int w    = tid >> 5;
  const int b    = blockIdx.y;
  const int mBase = blockIdx.x * 16;
  const int l15  = lane & 15;
  const int hi   = lane >> 4;
  const int klo8 = hi << 3;
  const int koff16 = hi << 4;

#pragma unroll
  for (int i = 0; i < 8; ++i) {            // 16 rows * 128 chunks
    int c   = tid + (i << 8);
    int r   = c >> 7;
    int off = (c & 127) << 3;
    cp16_g2l(&sQ[r * QP + off],
             Q + ((size_t)b * 2048 + mBase + r) * 1024 + off);
  }
  cp_wait();
  __syncthreads();

  v8f acc[16];
  const __bf16* Qa = &sQ[l15 * QP];
#pragma unroll
  for (int t = 0; t < 16; ++t) {
    int st = t * 8 + w;                     // s-tile 0..127
    v8f c = (v8f){0,0,0,0,0,0,0,0};
    const __bf16* Kb = Km + ((size_t)b * 2048 + st * 16 + l15) * 1024;
#pragma unroll 4
    for (int kt = 0; kt < 32; ++kt) {
      const __bf16* pa = Qa + kt * 32 + klo8;
      v16bf a  = cat88(*(const v8bf*)pa, *(const v8bf*)(pa + 16));
      const __bf16* pb = Kb + kt * 32 + klo8;
      v16bf bb = cat88(*(const v8bf*)pb, *(const v8bf*)(pb + 16));
      c = wmma_bf16(a, bb, c);
    }
#pragma unroll
    for (int r = 0; r < 8; ++r) c[r] *= scale;
    acc[t] = c;
  }

  // row max (rows r + 8*hi; cols striped over 16 lanes of each half + 8 waves)
  float rmax[8];
  {
    float mx[8];
#pragma unroll
    for (int r = 0; r < 8; ++r) {
      float v = -3.0e38f;
#pragma unroll
      for (int t = 0; t < 16; ++t) v = fmaxf(v, acc[t][r]);
#pragma unroll
      for (int d = 1; d < 16; d <<= 1) v = fmaxf(v, __shfl_xor(v, d, 32));
      mx[r] = v;
    }
    if (lane == 0 || lane == 16) {
#pragma unroll
      for (int r = 0; r < 8; ++r) sredM[w][hi * 8 + r] = mx[r];
    }
    __syncthreads();
#pragma unroll
    for (int r = 0; r < 8; ++r) {
      int row = hi * 8 + r;
      float v = sredM[0][row];
#pragma unroll
      for (int w2 = 1; w2 < 8; ++w2) v = fmaxf(v, sredM[w2][row]);
      rmax[r] = v;
    }
  }

  // exp + row sum
  float sm[8] = {0,0,0,0,0,0,0,0};
#pragma unroll
  for (int t = 0; t < 16; ++t)
#pragma unroll
    for (int r = 0; r < 8; ++r) {
      float e = __expf(acc[t][r] - rmax[r]);
      acc[t][r] = e;
      sm[r] += e;
    }
#pragma unroll
  for (int r = 0; r < 8; ++r) {
    float v = sm[r];
#pragma unroll
    for (int d = 1; d < 16; d <<= 1) v += __shfl_xor(v, d, 32);
    sm[r] = v;
  }
  if (lane == 0 || lane == 16) {
#pragma unroll
    for (int r = 0; r < 8; ++r) sredS[w][hi * 8 + r] = sm[r];
  }
  __syncthreads();
  float rinv[8];
#pragma unroll
  for (int r = 0; r < 8; ++r) {
    int row = hi * 8 + r;
    float s = 0.0f;
#pragma unroll
    for (int w2 = 0; w2 < 8; ++w2) s += sredS[w2][row];
    rinv[r] = 1.0f / s;
  }

  // write normalized probs (bf16)
  const size_t pbase = (size_t)b * 2048 * 2048;
#pragma unroll
  for (int t = 0; t < 16; ++t) {
    int st = t * 8 + w;
#pragma unroll
    for (int r = 0; r < 8; ++r) {
      int row = mBase + hi * 8 + r;
      P[pbase + (size_t)row * 2048 + st * 16 + l15] = (__bf16)(acc[t][r] * rinv[r]);
    }
  }
}

// ---------------- launcher ----------------

extern "C" void kernel_launch(void* const* d_in, const int* in_sizes, int n_in,
                              void* d_out, int out_size, void* d_ws, size_t ws_size,
                              hipStream_t stream) {
  (void)in_sizes; (void)n_in; (void)out_size; (void)ws_size;
  const float* x  = (const float*)d_in[0];
  const float* Wq = (const float*)d_in[1];
  const float* bq = (const float*)d_in[2];
  const float* Wk = (const float*)d_in[3];
  const float* bk = (const float*)d_in[4];
  const float* Wv = (const float*)d_in[5];
  const float* bv = (const float*)d_in[6];
  const float* Wo = (const float*)d_in[7];
  const float* bo = (const float*)d_in[8];

  const int Bb = 4, T = 2048, D = 1024, M = Bb * T;

  __bf16* wsb = (__bf16*)d_ws;
  size_t o = 0;
  __bf16* xb  = wsb + o; o += (size_t)M * D;     // x bf16
  __bf16* wqb = wsb + o; o += (size_t)D * D;
  __bf16* wkb = wsb + o; o += (size_t)D * D;
  __bf16* wvb = wsb + o; o += (size_t)D * D;
  __bf16* wob = wsb + o; o += (size_t)D * D;
  __bf16* qb  = wsb + o; o += (size_t)M * D;     // Q
  __bf16* kb  = wsb + o; o += (size_t)M * D;     // K
  __bf16* vtb = wsb + o; o += (size_t)M * D;     // V^T  [D x M]
  __bf16* pb  = wsb + o; o += (size_t)Bb * T * T;// probs
  __bf16* ob  = wsb + o; o += (size_t)M * D;     // attention output

  dim3 blk(256);
  cvt_f32_bf16<<<2048, blk, 0, stream>>>(x,  xb,  M * D / 4);
  cvt_f32_bf16<<<512,  blk, 0, stream>>>(Wq, wqb, D * D / 4);
  cvt_f32_bf16<<<512,  blk, 0, stream>>>(Wk, wkb, D * D / 4);
  cvt_f32_bf16<<<512,  blk, 0, stream>>>(Wv, wvb, D * D / 4);
  cvt_f32_bf16<<<512,  blk, 0, stream>>>(Wo, wob, D * D / 4);

  dim3 gP(D / BT, M / BT, 1);
  // Q = x Wq^T + bq ; K = x Wk^T + bk ; V^T (transposed store) = (x Wv^T + bv)^T
  gemm_nt_bf16<0><<<gP, blk, 0, stream>>>(xb, wqb, bq, qb, D, D, D, D, 0, 0, 0);
  gemm_nt_bf16<0><<<gP, blk, 0, stream>>>(xb, wkb, bk, kb, D, D, D, D, 0, 0, 0);
  gemm_nt_bf16<1><<<gP, blk, 0, stream>>>(xb, wvb, bv, vtb, D, D, D, M, 0, 0, 0);

  // probs = softmax(Q K^T / 32)
  attn_scores_softmax<<<dim3(T / 16, Bb), blk, 0, stream>>>(qb, kb, pb, 1.0f / 32.0f);

  // O = probs @ V   (B operand = V^T slice per batch -> NT layout)
  gemm_nt_bf16<0><<<dim3(D / BT, T / BT, Bb), blk, 0, stream>>>(
      pb, vtb, nullptr, ob, T, T, M, D,
      (long long)T * T, (long long)T, (long long)T * D);

  // y = O Wo^T + bo  (f32 out)
  gemm_nt_bf16<2><<<gP, blk, 0, stream>>>(ob, wob, bo, d_out, D, D, D, D, 0, 0, 0);
}
